// TransitionDown_35656818492086
// MI455X (gfx1250) — matrix operand order repaired and checked by reading, not compiled, not run on hardware
//
#include <hip/hip_runtime.h>
#include <stdint.h>

// ---------------- problem constants ----------------
#define BATCH   16
#define NPTS    4096
#define DF      128
#define NGRP    1024
#define KNN     16
#define DOUT    256
#define KPAD    160           // padded K for WMMA (coords at 0..2, feats 8..135, rest 0)
#define KSTRIDE 168           // global Wh row stride in halves
#define WCH_STR 40            // LDS W-chunk col stride in halves (80B = 20 dwords)
#define MTOT    (BATCH*NGRP*KNN)   // 262144 GEMM rows
#define NBLK    (MTOT/64)          // 4096 GEMM blocks
#define FEAT_OUT_ELEMS (BATCH*NGRP*DOUT)
#define BN_EPS  1e-5f

typedef __attribute__((ext_vector_type(16))) _Float16 v16h;
typedef __attribute__((ext_vector_type(8)))  _Float16 v8h;
typedef __attribute__((ext_vector_type(4)))  _Float16 v4h;
typedef __attribute__((ext_vector_type(8)))  float    v8f;
typedef __attribute__((ext_vector_type(4)))  unsigned int v4u;
typedef __attribute__((ext_vector_type(8)))  int      v8i_t;
typedef __attribute__((ext_vector_type(4)))  int      v4i_t;

static __device__ __forceinline__ v16h cat8(v8h lo, v8h hi) {
  return __builtin_shufflevector(lo, hi, 0,1,2,3,4,5,6,7,8,9,10,11,12,13,14,15);
}

// TDM: async-load a [256 rows x 32 halves] tile of Wh (row stride 168 halves)
// into LDS with 16B padding after each 64B row => LDS col stride 40 halves.
static __device__ __forceinline__ void tdm_load_wchunk(const _Float16* gsrc,
                                                       _Float16* ldst) {
  unsigned long long ga = (unsigned long long)(uintptr_t)gsrc;
  unsigned int lds = (unsigned int)(uintptr_t)ldst;  // low 32 bits = LDS byte addr
  // group0: count=1 (valid user D#), lds_addr, 57b global addr, type=2 (image)
  v4u g0 = { 1u, lds, (unsigned int)ga,
             (unsigned int)((ga >> 32) & 0x1FFFFFFull) | (2u << 30) };
  // group1: data_size=2B, pad_enable, pad_interval=16 DW (64B), pad_amount=4 DW (16B)
  //         tensor_dim0=32, tensor_dim1=256, tile_dim0=32, tile_dim1=256,
  //         tensor_dim0_stride=168
  v8i_t g1 = { (int)((1u << 16) | (1u << 20) | (3u << 22) | (3u << 25)),
               (int)(32u << 16),     // [47:32]=atomic_bar=0, [63:48]=dim0.lo
               (int)(256u << 16),    // [79:64]=dim0.hi=0,  [95:80]... dim1.lo
               (int)(32u << 16),     // [111:96]=dim1.hi=0, [127:112]=tile_dim0
               256,                  // [143:128]=tile_dim1, [159:144]=tile_dim2=0
               168,                  // [191:160]=dim0_stride.lo
               0, 0 };
  v4i_t gz = {0, 0, 0, 0};
#if __clang_major__ >= 23
  v8i_t gz8 = {0, 0, 0, 0, 0, 0, 0, 0};
  __builtin_amdgcn_tensor_load_to_lds(g0, g1, gz, gz, gz8, 0);
#else
  __builtin_amdgcn_tensor_load_to_lds(g0, g1, gz, gz, 0);
#endif
}

// ---------------- 1) farthest point sampling (one 1024-thread WG per batch) ----
__global__ __launch_bounds__(1024) void td_fps_kernel(const float* __restrict__ coord,
                                                      float* __restrict__ center_coord) {
  __shared__ float xs[NPTS], ys[NPTS], zs[NPTS];
  __shared__ float rv[32];
  __shared__ int   ri[32];
  __shared__ int   s_cur;
  const int b = blockIdx.x;
  const int tid = threadIdx.x;

  #pragma unroll
  for (int t = 0; t < 4; ++t) {
    int p = tid + 1024 * t;
    const float* cp = coord + ((size_t)b * NPTS + p) * 3;
    xs[p] = cp[0]; ys[p] = cp[1]; zs[p] = cp[2];
  }
  float dist[4];
  #pragma unroll
  for (int t = 0; t < 4; ++t) dist[t] = __builtin_inff();
  __syncthreads();

  int cur = 0;
  if (tid == 0) {
    float* cc = center_coord + ((size_t)b * NGRP) * 3;
    cc[0] = xs[0]; cc[1] = ys[0]; cc[2] = zs[0];
  }

  for (int g = 1; g < NGRP; ++g) {
    const float cx = xs[cur], cy = ys[cur], cz = zs[cur];
    float bestv = -1.0f; int besti = 0;
    #pragma unroll
    for (int t = 0; t < 4; ++t) {
      int p = tid + 1024 * t;
      float dx = xs[p] - cx, dy = ys[p] - cy, dz = zs[p] - cz;
      float d = dx * dx + dy * dy + dz * dz;
      dist[t] = fminf(dist[t], d);
      if (dist[t] > bestv || (dist[t] == bestv && p < besti)) { bestv = dist[t]; besti = p; }
    }
    // wave32 reduce (prefer larger value, tie -> smaller index, matching argmax)
    #pragma unroll
    for (int off = 16; off > 0; off >>= 1) {
      float ov = __shfl_xor(bestv, off);
      int   oi = __shfl_xor(besti, off);
      if (ov > bestv || (ov == bestv && oi < besti)) { bestv = ov; besti = oi; }
    }
    if ((tid & 31) == 0) { rv[tid >> 5] = bestv; ri[tid >> 5] = besti; }
    __syncthreads();
    if (tid < 32) {
      float v = rv[tid]; int i = ri[tid];
      #pragma unroll
      for (int off = 16; off > 0; off >>= 1) {
        float ov = __shfl_xor(v, off);
        int   oi = __shfl_xor(i, off);
        if (ov > v || (ov == v && oi < i)) { v = ov; i = oi; }
      }
      if (tid == 0) {
        s_cur = i;
        float* cc = center_coord + ((size_t)b * NGRP + g) * 3;
        cc[0] = xs[i]; cc[1] = ys[i]; cc[2] = zs[i];
      }
    }
    __syncthreads();
    cur = s_cur;
  }
}

// ---------------- 2) kNN: one thread per center, register insertion sort -------
__global__ __launch_bounds__(256) void td_knn_kernel(const float* __restrict__ coord,
                                                     const float* __restrict__ center_coord,
                                                     int* __restrict__ neigh_idx) {
  const int c = blockIdx.x * 256 + threadIdx.x;   // [0, BATCH*NGRP)
  const int b = c >> 10;
  const float* cc = center_coord + (size_t)c * 3;
  const float cx = cc[0], cy = cc[1], cz = cc[2];

  float hd[KNN]; int hi[KNN];
  #pragma unroll
  for (int i = 0; i < KNN; ++i) { hd[i] = 3.4e38f; hi[i] = 0; }

  const float* base = coord + (size_t)b * NPTS * 3;
  for (int j = 0; j < NPTS; ++j) {
    const float* p = base + (size_t)j * 3;
    float dx = p[0] - cx, dy = p[1] - cy, dz = p[2] - cz;
    float d = dx * dx + dy * dy + dz * dz;
    if (d < hd[KNN - 1]) {
      hd[KNN - 1] = d; hi[KNN - 1] = j;
      #pragma unroll
      for (int i = KNN - 1; i > 0; --i) {
        if (hd[i] < hd[i - 1]) {
          float td = hd[i]; hd[i] = hd[i - 1]; hd[i - 1] = td;
          int   ti = hi[i]; hi[i] = hi[i - 1]; hi[i - 1] = ti;
        }
      }
    }
  }
  #pragma unroll
  for (int k = 0; k < KNN; ++k) neigh_idx[(size_t)c * KNN + k] = hi[k];
}

// ---------------- 3) repack W (f32 [256][131]) -> f16 [256][KSTRIDE] ----------
__global__ __launch_bounds__(256) void td_wcvt_kernel(const float* __restrict__ W,
                                                      _Float16* __restrict__ Wh) {
  int id = blockIdx.x * 256 + threadIdx.x;
  if (id >= DOUT * KSTRIDE) return;
  int n = id / KSTRIDE, kp = id % KSTRIDE;
  float v = 0.0f;
  if (kp < 3)                  v = W[n * 131 + kp];        // relative-coord part
  else if (kp >= 8 && kp < 136) v = W[n * 131 + (kp - 5)]; // feature part
  Wh[id] = (_Float16)v;
}

// ---------------- 4) fused gather + WMMA GEMM + BN partial stats + min/max ----
// block = 256 threads (8 waves), covers 64 rows (= 4 groups of K=16) x 256 cols.
// W chunks are TDM-DMA'd into a double-buffered, pad-strided LDS tile while the
// previous chunk feeds the WMMA pipeline.
__global__ __launch_bounds__(256) void td_gemm_kernel(const float* __restrict__ feature,
                                                      const float* __restrict__ coord,
                                                      const float* __restrict__ center_coord,
                                                      const int* __restrict__ neigh_idx,
                                                      const _Float16* __restrict__ Whg,
                                                      float* __restrict__ ymax,
                                                      float* __restrict__ ymin,
                                                      float* __restrict__ psum,
                                                      float* __restrict__ psq) {
  __shared__ __align__(16) char smem[64 * KSTRIDE * 2 + 2 * DOUT * WCH_STR * 2];
  _Float16* As  = (_Float16*)smem;                            // 64 x 168 halves
  _Float16* Wc0 = (_Float16*)(smem + 64 * KSTRIDE * 2);       // 256 x 40 halves
  _Float16* Wc1 = Wc0 + DOUT * WCH_STR;                       // 256 x 40 halves
  float* s_sum = (float*)smem;                                // aliases As later
  float* s_sq  = (float*)(smem + 1024);

  const int tid  = threadIdx.x;
  const int wave = tid >> 5, lane = tid & 31;

  // kick off DMA of W chunk 0 immediately; it overlaps the A-tile gather
  if (wave == 0) tdm_load_wchunk(Whg, Wc0);

  // ---- gather A tile: 4 threads per row ----
  {
    const int r = tid >> 2, t4 = tid & 3;
    const int m  = blockIdx.x * 64 + r;
    const int bb = m >> 14;
    const int gk = m & 16383;
    const int gg = gk >> 4;
    const int idx = neigh_idx[m];
    _Float16* ar = As + r * KSTRIDE;
    const float* fp = feature + ((size_t)bb * NPTS + idx) * DF + t4 * 32;
    #pragma unroll
    for (int i = 0; i < 8; ++i) {
      float4 v = ((const float4*)fp)[i];
      v4h h = {(_Float16)v.x, (_Float16)v.y, (_Float16)v.z, (_Float16)v.w};
      *(v4h*)(ar + 8 + t4 * 32 + 4 * i) = h;
    }
    v4h z = {};
    *(v4h*)(ar + 136 + t4 * 8)     = z;   // tail padding 136..167
    *(v4h*)(ar + 136 + t4 * 8 + 4) = z;
    if (t4 == 0) {
      const float* pc = coord        + ((size_t)bb * NPTS + idx) * 3;
      const float* cc = center_coord + ((size_t)bb * NGRP + gg) * 3;
      ar[0] = (_Float16)(pc[0] - cc[0]);
      ar[1] = (_Float16)(pc[1] - cc[1]);
      ar[2] = (_Float16)(pc[2] - cc[2]);
      ar[3] = (_Float16)0.f; ar[4] = (_Float16)0.f; ar[5] = (_Float16)0.f;
      ar[6] = (_Float16)0.f; ar[7] = (_Float16)0.f;
    }
  }

  const int strip = wave >> 1;           // row group of 16 (== one (b,g) group)
  const int colhalf = wave & 1;          // 128-col half
  const int lane15 = lane & 15;
  const int khalf = lane >> 4;
  const int rowbase = strip * 16 + lane15;

  v8f acc[8];
  #pragma unroll
  for (int t = 0; t < 8; ++t)
    #pragma unroll
    for (int j = 0; j < 8; ++j) acc[t][j] = 0.0f;

  // ---- K loop: TDM double-buffered W chunks, 8 WMMA tiles per wave per step --
  for (int step = 0; step < 5; ++step) {
    const int kk = step * 32;
    _Float16* Wcur = (step & 1) ? Wc1 : Wc0;
    if (wave == 0) {
      if (step + 1 < 5) {
        _Float16* Wnxt = ((step + 1) & 1) ? Wc1 : Wc0;
        tdm_load_wchunk(Whg + (kk + 32), Wnxt);       // prefetch next chunk
        __builtin_amdgcn_s_wait_tensorcnt(1);         // current chunk landed
      } else {
        __builtin_amdgcn_s_wait_tensorcnt(0);
      }
    }
    __syncthreads();

    // A fragment per ISA 16-bit 16x32 layout
    v8h alo = *(const v8h*)(As + rowbase * KSTRIDE + kk + khalf * 8);
    v8h ahi = *(const v8h*)(As + rowbase * KSTRIDE + kk + 16 + khalf * 8);
    v16h a = cat8(alo, ahi);

    // preload all 8 B fragments, then run the WMMA chain without LDS stalls
    v16h bf[8];
    #pragma unroll
    for (int t = 0; t < 8; ++t) {
      int col = colhalf * 128 + t * 16 + lane15;
      v8h blo = *(const v8h*)(Wcur + col * WCH_STR + khalf * 8);
      v8h bhi = *(const v8h*)(Wcur + col * WCH_STR + 16 + khalf * 8);
      bf[t] = cat8(blo, bhi);
    }
    #pragma unroll
    for (int t = 0; t < 8; ++t) {
      acc[t] = __builtin_amdgcn_wmma_f32_16x16x32_f16(
          false, a, false, bf[t], (short)0, acc[t], false, false);
    }
    __syncthreads();
  }

  // ---- min/max over K (one strip == one group) + per-channel partial stats ----
  s_sum[tid] = 0.0f;
  s_sq[tid]  = 0.0f;
  __syncthreads();

  const int grp = blockIdx.x * 4 + strip;
  #pragma unroll
  for (int t = 0; t < 8; ++t) {
    float mx = acc[t][0], mn = acc[t][0], s = 0.0f, q = 0.0f;
    #pragma unroll
    for (int j = 0; j < 8; ++j) {
      float v = acc[t][j];
      mx = fmaxf(mx, v); mn = fminf(mn, v);
      s += v; q += v * v;
    }
    int col = colhalf * 128 + t * 16 + lane15;
    atomicAdd(&s_sum[col], s);
    atomicAdd(&s_sq[col], q);
    mx = fmaxf(mx, __shfl_xor(mx, 16));
    mn = fminf(mn, __shfl_xor(mn, 16));
    if (lane < 16) {
      ymax[(size_t)grp * DOUT + col] = mx;
      ymin[(size_t)grp * DOUT + col] = mn;
    }
  }
  __syncthreads();
  psum[(size_t)blockIdx.x * DOUT + tid] = s_sum[tid];
  psq [(size_t)blockIdx.x * DOUT + tid] = s_sq[tid];
}

// ---------------- 5) reduce BN partials -> mean / rstd per channel ------------
__global__ __launch_bounds__(256) void td_stats_kernel(const float* __restrict__ psum,
                                                       const float* __restrict__ psq,
                                                       float* __restrict__ mr) {
  __shared__ float ss[256], sq[256];
  const int ch = blockIdx.x, tid = threadIdx.x;
  float s = 0.0f, q = 0.0f;
  #pragma unroll
  for (int i = 0; i < NBLK / 256; ++i) {
    int p = tid + 256 * i;
    s += psum[(size_t)p * DOUT + ch];
    q += psq [(size_t)p * DOUT + ch];
  }
  ss[tid] = s; sq[tid] = q;
  __syncthreads();
  #pragma unroll
  for (int off = 128; off > 0; off >>= 1) {
    if (tid < off) { ss[tid] += ss[tid + off]; sq[tid] += sq[tid + off]; }
    __syncthreads();
  }
  if (tid == 0) {
    const float inv_cnt = 1.0f / (float)MTOT;
    float mean = ss[0] * inv_cnt;
    float var  = sq[0] * inv_cnt - mean * mean;
    var = fmaxf(var, 0.0f);
    mr[ch]        = mean;
    mr[DOUT + ch] = rsqrtf(var + BN_EPS);
  }
}

// ---------------- 6) finalize: BN affine + ReLU on max/min --------------------
__global__ __launch_bounds__(256) void td_final_kernel(const float* __restrict__ ymax,
                                                       const float* __restrict__ ymin,
                                                       const float* __restrict__ mr,
                                                       const float* __restrict__ gamma,
                                                       const float* __restrict__ beta,
                                                       float* __restrict__ out) {
  const size_t id = (size_t)blockIdx.x * 256 + threadIdx.x;
  const int o = (int)(id & (DOUT - 1));
  float m = mr[o], rs = mr[DOUT + o];
  float s = gamma[o] * rs;
  float v = (s >= 0.0f) ? ymax[id] : ymin[id];   // monotone => max commutes with BN+ReLU
  float r = (v - m) * s + beta[o];
  out[id] = r > 0.0f ? r : 0.0f;
}

// ---------------- launcher ----------------------------------------------------
extern "C" void kernel_launch(void* const* d_in, const int* in_sizes, int n_in,
                              void* d_out, int out_size, void* d_ws, size_t ws_size,
                              hipStream_t stream) {
  (void)in_sizes; (void)n_in; (void)out_size; (void)ws_size;
  const float* feature = (const float*)d_in[0];
  const float* coord   = (const float*)d_in[1];
  const float* W       = (const float*)d_in[2];
  // d_in[3] (bias) is mathematically absorbed by train-mode BatchNorm
  const float* gamma   = (const float*)d_in[4];
  const float* beta    = (const float*)d_in[5];

  float* out_feat  = (float*)d_out;
  float* out_coord = out_feat + (size_t)FEAT_OUT_ELEMS;

  char* ws = (char*)d_ws;
  size_t off = 0;
  _Float16* Wh = (_Float16*)(ws + off); off += (size_t)DOUT * KSTRIDE * 2;   // 86016
  float* ymax  = (float*)(ws + off);    off += (size_t)BATCH * NGRP * DOUT * 4;
  float* ymin  = (float*)(ws + off);    off += (size_t)BATCH * NGRP * DOUT * 4;
  float* psum  = (float*)(ws + off);    off += (size_t)NBLK * DOUT * 4;
  float* psq   = (float*)(ws + off);    off += (size_t)NBLK * DOUT * 4;
  float* mr    = (float*)(ws + off);    off += 2 * DOUT * 4;
  int* nidx    = (int*)(ws + off);      off += (size_t)MTOT * 4;

  td_fps_kernel<<<BATCH, 1024, 0, stream>>>(coord, out_coord);
  td_knn_kernel<<<(BATCH * NGRP) / 256, 256, 0, stream>>>(coord, out_coord, nidx);
  td_wcvt_kernel<<<(DOUT * KSTRIDE + 255) / 256, 256, 0, stream>>>(W, Wh);
  td_gemm_kernel<<<NBLK, 256, 0, stream>>>(feature, coord, out_coord, nidx, Wh,
                                           ymax, ymin, psum, psq);
  td_stats_kernel<<<DOUT, 256, 0, stream>>>(psum, psq, mr);
  td_final_kernel<<<(BATCH * NGRP * DOUT) / 256, 256, 0, stream>>>(
      ymax, ymin, mr, gamma, beta, out_feat);
}
